// Oracle850BTransformer_50697793962047
// MI455X (gfx1250) — matrix operand order repaired and checked by reading, not compiled, not run on hardware
//
#include <hip/hip_runtime.h>
#include <hip/hip_bf16.h>

// ---------------------------------------------------------------------------
// MoE transformer forward for MI455X (gfx1250, wave32, WMMA bf16 path).
// All GEMMs: C = A(bf16)[M,K] * B(bf16)[N,K]^T via v_wmma_f32_16x16x32_bf16.
// ---------------------------------------------------------------------------

typedef __bf16 bf16_t;
typedef __attribute__((ext_vector_type(16))) __bf16 v16bf;
typedef __attribute__((ext_vector_type(8)))  float  v8f;

union FragAB { v16bf v; uint4 u[2]; };
union AccU   { v8f  v; float f[8]; };

constexpr int TTOK   = 1024;   // B*S tokens
constexpr int DMODEL = 1024;
constexpr int NHEAD  = 16;
constexpr int DHEAD  = 64;
constexpr int DFFN   = 2048;
constexpr int NEXP   = 8;
constexpr int NLAY   = 2;
constexpr int VOC    = 32000;
constexpr int GROWS  = 3072;   // MoE routed rows, 128-aligned per expert (<= 2048 + 8*127)

#define GEMM_BM 128
#define GEMM_BN 128
#define GEMM_BK 32
#define LDS_PK  40   // bf16 row stride in LDS (80B: 16B aligned, bank-spread)

// ---------------------------------------------------------------------------
// Tiled bf16 WMMA GEMM. 256 threads = 8 waves (2x4 wave grid), 128x128 tile.
// Double-buffered LDS (one barrier per K step), K-invariant addressing hoisted
// out of the main loop (branchless gather), global prefetch 2 tiles ahead.
//   - z-batched via sA/sB/sC element strides (attention heads)
//   - optional gatherA: A row index map (MoE token gather), -1 => zero row
//   - optional scatterC: C row index map (MoE slot scatter), -1 => skip store
//   - optional tileExpert + sBe: per-M-tile B base offset (MoE expert weights)
// ---------------------------------------------------------------------------
__global__ __launch_bounds__(256)
void k_gemm_bf16(const bf16_t* __restrict__ A, const bf16_t* __restrict__ B,
                 float* __restrict__ C, const float* __restrict__ bias,
                 int M, int N, int Kd, int lda, int ldb, int ldc,
                 long long sA, long long sB, long long sC, float alpha,
                 const int* __restrict__ gatherA, const int* __restrict__ scatterC,
                 const int* __restrict__ tileExpert, long long sBe)
{
    __shared__ bf16_t As[2][GEMM_BM * LDS_PK];
    __shared__ bf16_t Bs[2][GEMM_BN * LDS_PK];

    const int tid = threadIdx.x;
    const int tn = blockIdx.x, tm = blockIdx.y, z = blockIdx.z;
    const bf16_t* Ab = A + (size_t)z * sA;
    const bf16_t* Bb = B + (size_t)z * sB +
                       (tileExpert ? (size_t)tileExpert[tm] * sBe : 0);
    float* Cb = C + (size_t)z * sC;

    const int wid = tid >> 5, lane = tid & 31;
    const int wm = wid & 1, wn = wid >> 1;   // 2x4 wave grid
    const int lr = lane & 15, lh = lane >> 4;

    // ---- K-invariant per-thread addresses for cooperative tile loads ------
    // A/B tile: 128 rows x 32 K bf16 = 512 x 16B chunks; thread covers 2.
    const bf16_t* aPtr[2]; const bf16_t* bPtr[2];
    bool aOK[2], bOK[2];
    int  sOff[2];
    #pragma unroll
    for (int i = 0; i < 2; ++i) {
        int idx = tid + i * 256;
        int row = idx >> 2;
        int seg = (idx & 3) * 8;            // bf16 elems within the K=32 row
        sOff[i] = row * LDS_PK + seg;
        int am = tm * GEMM_BM + row;
        bool ok = (am < M);
        int src = ok ? am : 0;
        if (gatherA) {
            int g = ok ? gatherA[am] : -1;  // one-time divergent index load
            ok = (g >= 0);
            src = ok ? g : 0;
        }
        aOK[i] = ok;
        aPtr[i] = Ab + (size_t)src * lda + seg;
        int bn = tn * GEMM_BN + row;
        bOK[i] = (bn < N);
        bPtr[i] = Bb + (size_t)(bOK[i] ? bn : 0) * ldb + seg;
    }

    // ---- K-invariant LDS fragment offsets ---------------------------------
    int faOff[4], fbOff[2];
    #pragma unroll
    for (int i = 0; i < 4; ++i) faOff[i] = (wm * 64 + i * 16 + lr) * LDS_PK + lh * 8;
    #pragma unroll
    for (int j = 0; j < 2; ++j) fbOff[j] = (wn * 32 + j * 16 + lr) * LDS_PK + lh * 8;

    v8f acc[4][2] = {};
    const int nk = Kd / GEMM_BK;

    uint4 ra[2], rb[2];
    auto fetch = [&](int kt) {              // branchless: load row0 + select 0
        #pragma unroll
        for (int i = 0; i < 2; ++i) {
            uint4 va = *(const uint4*)(aPtr[i] + (size_t)kt * GEMM_BK);
            uint4 vb = *(const uint4*)(bPtr[i] + (size_t)kt * GEMM_BK);
            ra[i] = aOK[i] ? va : make_uint4(0u, 0u, 0u, 0u);
            rb[i] = bOK[i] ? vb : make_uint4(0u, 0u, 0u, 0u);
        }
    };
    auto stage = [&](int buf) {
        #pragma unroll
        for (int i = 0; i < 2; ++i) {
            *(uint4*)(&As[buf][sOff[i]]) = ra[i];
            *(uint4*)(&Bs[buf][sOff[i]]) = rb[i];
        }
    };

    fetch(0);
    stage(0);
    __syncthreads();

    for (int kt = 0; kt < nk; ++kt) {
        const int cur = kt & 1;
        const bool more = (kt + 1 < nk);
        if (more) {
            fetch(kt + 1);                  // global prefetch overlaps WMMA
            if (kt + 2 < nk) {              // L2 prefetch 2 tiles ahead
                __builtin_prefetch(aPtr[0] + (size_t)(kt + 2) * GEMM_BK, 0, 3);
                __builtin_prefetch(bPtr[0] + (size_t)(kt + 2) * GEMM_BK, 0, 3);
            }
        }

        // A fragment (16x32): lane holds row lane%16; K chunks [8h,+8),[16+8h,+8)
        FragAB fa[4], fb[2];
        #pragma unroll
        for (int i = 0; i < 4; ++i) {
            fa[i].u[0] = *(const uint4*)(&As[cur][faOff[i]]);
            fa[i].u[1] = *(const uint4*)(&As[cur][faOff[i] + 16]);
        }
        // B fragment (32x16): lane holds col lane%16; K chunks [8h,+8),[16+8h,+8)
        #pragma unroll
        for (int j = 0; j < 2; ++j) {
            fb[j].u[0] = *(const uint4*)(&Bs[cur][fbOff[j]]);
            fb[j].u[1] = *(const uint4*)(&Bs[cur][fbOff[j] + 16]);
        }
        #pragma unroll
        for (int i = 0; i < 4; ++i)
            #pragma unroll
            for (int j = 0; j < 2; ++j)
                acc[i][j] = __builtin_amdgcn_wmma_f32_16x16x32_bf16(
                    false, fa[i].v, false, fb[j].v, (short)0, acc[i][j], false, false);

        if (more) {
            stage(cur ^ 1);                 // safe: last read of buf^1 retired
            __syncthreads();                //   at the previous barrier
        }
    }

    // C layout: lane col = lane%16, VGPR r holds M = 8*(lane/16)+r
    #pragma unroll
    for (int j = 0; j < 2; ++j) {
        int c = tn * GEMM_BN + wn * 32 + j * 16 + lr;
        if (c >= N) continue;
        float bv = bias ? bias[c] : 0.f;
        #pragma unroll
        for (int i = 0; i < 4; ++i) {
            AccU u; u.v = acc[i][j];
            int mb = tm * GEMM_BM + wm * 64 + i * 16 + lh * 8;
            #pragma unroll
            for (int r = 0; r < 8; ++r) {
                int m = mb + r;
                if (m >= M) continue;
                long long cm = m;
                if (scatterC) { int sc = scatterC[m]; if (sc < 0) continue; cm = sc; }
                Cb[cm * (long long)ldc + c] = alpha * u.f[r] + bv;
            }
        }
    }
}

// ---------------------------------------------------------------------------
// Elementwise / small kernels
// ---------------------------------------------------------------------------
__global__ __launch_bounds__(256)
void k_embed(const int* __restrict__ ids, const float* __restrict__ tok,
             const float* __restrict__ pos, float* __restrict__ x)
{
    int t = blockIdx.x, d = threadIdx.x * 4;
    int id = ids[t];
    float4 a = *(const float4*)(tok + (size_t)id * DMODEL + d);
    float4 b = *(const float4*)(pos + (size_t)t * DMODEL + d);
    float4 r; r.x = a.x + b.x; r.y = a.y + b.y; r.z = a.z + b.z; r.w = a.w + b.w;
    *(float4*)(x + (size_t)t * DMODEL + d) = r;
}

__global__ __launch_bounds__(256)
void k_cvt(const float* __restrict__ in, bf16_t* __restrict__ out, long long n)
{
    long long i = ((long long)blockIdx.x * 256 + threadIdx.x) * 4;
    if (i + 3 < n) {
        float4 v = *(const float4*)(in + i);
        out[i]     = (bf16_t)v.x; out[i + 1] = (bf16_t)v.y;
        out[i + 2] = (bf16_t)v.z; out[i + 3] = (bf16_t)v.w;
    } else {
        for (long long j = i; j < n; ++j) out[j] = (bf16_t)in[j];
    }
}

// vT[(h*64+d)][t] = qkv[t][2D + h*64 + d]  (V^T per head, bf16)
__global__ __launch_bounds__(256)
void k_vtrans(const float* __restrict__ qkv, bf16_t* __restrict__ vT)
{
    int t = blockIdx.x;
    #pragma unroll
    for (int i = 0; i < 4; ++i) {
        int hd = threadIdx.x + i * 256;
        vT[(size_t)hd * TTOK + t] =
            (bf16_t)qkv[(size_t)t * (3 * DMODEL) + 2 * DMODEL + hd];
    }
}

__global__ __launch_bounds__(256)
void k_softmax1024(float* __restrict__ s)
{
    __shared__ float red[256];
    float* row = s + (size_t)blockIdx.x * 1024;
    int t = threadIdx.x;
    float v0 = row[t*4], v1 = row[t*4+1], v2 = row[t*4+2], v3 = row[t*4+3];
    float m = fmaxf(fmaxf(v0, v1), fmaxf(v2, v3));
    red[t] = m; __syncthreads();
    for (int o = 128; o > 0; o >>= 1) { if (t < o) red[t] = fmaxf(red[t], red[t+o]); __syncthreads(); }
    m = red[0]; __syncthreads();
    v0 = __expf(v0 - m); v1 = __expf(v1 - m); v2 = __expf(v2 - m); v3 = __expf(v3 - m);
    red[t] = v0 + v1 + v2 + v3; __syncthreads();
    for (int o = 128; o > 0; o >>= 1) { if (t < o) red[t] += red[t+o]; __syncthreads(); }
    float inv = 1.f / red[0];
    row[t*4] = v0*inv; row[t*4+1] = v1*inv; row[t*4+2] = v2*inv; row[t*4+3] = v3*inv;
}

// out = LN(a + (b?b:0)) * g + be   (in-place safe: each block touches its row)
__global__ __launch_bounds__(256)
void k_add_ln(const float* __restrict__ a, const float* __restrict__ b,
              const float* __restrict__ g, const float* __restrict__ be,
              float* __restrict__ out)
{
    __shared__ float red[256];
    int tid = threadIdx.x;
    size_t base = (size_t)blockIdx.x * DMODEL + tid * 4;
    float v[4];
    #pragma unroll
    for (int i = 0; i < 4; ++i) v[i] = a[base + i] + (b ? b[base + i] : 0.f);
    red[tid] = v[0] + v[1] + v[2] + v[3]; __syncthreads();
    for (int o = 128; o > 0; o >>= 1) { if (tid < o) red[tid] += red[tid+o]; __syncthreads(); }
    float mean = red[0] * (1.f / DMODEL); __syncthreads();
    float q = 0.f;
    #pragma unroll
    for (int i = 0; i < 4; ++i) { float d = v[i] - mean; q += d * d; }
    red[tid] = q; __syncthreads();
    for (int o = 128; o > 0; o >>= 1) { if (tid < o) red[tid] += red[tid+o]; __syncthreads(); }
    float rstd = rsqrtf(red[0] * (1.f / DMODEL) + 1e-5f);
    int d0 = tid * 4;
    #pragma unroll
    for (int i = 0; i < 4; ++i)
        out[base + i] = (v[i] - mean) * rstd * g[d0 + i] + be[d0 + i];
}

// x = LN(x + p0*y2[2t] + p1*y2[2t+1]) * g + be
__global__ __launch_bounds__(256)
void k_moe_ln(const float* __restrict__ x, const float* __restrict__ y2,
              const float* __restrict__ topp, const float* __restrict__ g,
              const float* __restrict__ be, float* __restrict__ out)
{
    __shared__ float red[256];
    int t = blockIdx.x, tid = threadIdx.x;
    float p0 = topp[t * 2], p1 = topp[t * 2 + 1];
    size_t bx = (size_t)t * DMODEL + tid * 4;
    size_t b0 = (size_t)(t * 2)     * DMODEL + tid * 4;
    size_t b1 = (size_t)(t * 2 + 1) * DMODEL + tid * 4;
    float v[4];
    #pragma unroll
    for (int i = 0; i < 4; ++i) v[i] = x[bx + i] + p0 * y2[b0 + i] + p1 * y2[b1 + i];
    red[tid] = v[0] + v[1] + v[2] + v[3]; __syncthreads();
    for (int o = 128; o > 0; o >>= 1) { if (tid < o) red[tid] += red[tid+o]; __syncthreads(); }
    float mean = red[0] * (1.f / DMODEL); __syncthreads();
    float q = 0.f;
    #pragma unroll
    for (int i = 0; i < 4; ++i) { float d = v[i] - mean; q += d * d; }
    red[tid] = q; __syncthreads();
    for (int o = 128; o > 0; o >>= 1) { if (tid < o) red[tid] += red[tid+o]; __syncthreads(); }
    float rstd = rsqrtf(red[0] * (1.f / DMODEL) + 1e-5f);
    int d0 = tid * 4;
    #pragma unroll
    for (int i = 0; i < 4; ++i)
        out[bx + i] = (v[i] - mean) * rstd * g[d0 + i] + be[d0 + i];
}

// per-token router: 8 dots over D, top-2 + softmax(top vals)
__global__ __launch_bounds__(256)
void k_router(const float* __restrict__ x, const float* __restrict__ rw,
              int* __restrict__ topidx, float* __restrict__ topp)
{
    __shared__ float red[256];
    __shared__ float lg[NEXP];
    int t = blockIdx.x, tid = threadIdx.x;
    float acc[NEXP] = {};
    #pragma unroll
    for (int i = 0; i < 4; ++i) {
        int d = tid * 4 + i;
        float xv = x[(size_t)t * DMODEL + d];
        #pragma unroll
        for (int e = 0; e < NEXP; ++e) acc[e] += xv * rw[(size_t)e * DMODEL + d];
    }
    for (int e = 0; e < NEXP; ++e) {
        red[tid] = acc[e]; __syncthreads();
        for (int o = 128; o > 0; o >>= 1) { if (tid < o) red[tid] += red[tid+o]; __syncthreads(); }
        if (tid == 0) lg[e] = red[0];
        __syncthreads();
    }
    if (tid == 0) {
        int i0 = 0;
        for (int e = 1; e < NEXP; ++e) if (lg[e] > lg[i0]) i0 = e;
        int i1 = -1;
        for (int e = 0; e < NEXP; ++e)
            if (e != i0 && (i1 < 0 || lg[e] > lg[i1])) i1 = e;
        float e0 = 1.f, e1 = __expf(lg[i1] - lg[i0]);   // max-sub softmax over top2
        float inv = 1.f / (e0 + e1);
        topidx[t * 2] = i0; topidx[t * 2 + 1] = i1;
        topp[t * 2] = e0 * inv; topp[t * 2 + 1] = e1 * inv;
    }
}

// deterministic expert list build (single thread; ~5K trivial iterations)
__global__ void k_build(const int* __restrict__ topidx, int* __restrict__ tokmap,
                        int* __restrict__ rowmap, int* __restrict__ tileE)
{
    if (threadIdx.x != 0 || blockIdx.x != 0) return;
    int cnt[NEXP]; for (int e = 0; e < NEXP; ++e) cnt[e] = 0;
    for (int t = 0; t < TTOK; ++t) { cnt[topidx[2*t]]++; cnt[topidx[2*t+1]]++; }
    for (int g = 0; g < GROWS; ++g) { tokmap[g] = -1; rowmap[g] = -1; }
    for (int ti = 0; ti < GROWS / 128; ++ti) tileE[ti] = 0;
    int pos[NEXP]; int cur = 0;
    for (int e = 0; e < NEXP; ++e) {
        pos[e] = cur;
        int end  = cur + cnt[e];
        int aend = (end + 127) & ~127;
        for (int ti = cur >> 7; ti < (aend >> 7); ++ti) tileE[ti] = e;
        cur = aend;
    }
    for (int t = 0; t < TTOK; ++t)
        for (int k = 0; k < 2; ++k) {
            int e = topidx[2 * t + k];
            int g = pos[e]++;
            tokmap[g] = t;             // A gather: token row
            rowmap[g] = t * 2 + k;     // C scatter: (token,slot) row
        }
}

__global__ __launch_bounds__(256)
void k_silu_mul(const float* __restrict__ h1, const float* __restrict__ h3,
                bf16_t* __restrict__ act)
{
    size_t i = ((size_t)blockIdx.x * 256 + threadIdx.x) * 4;
    float4 a = *(const float4*)(h1 + i);
    float4 b = *(const float4*)(h3 + i);
    act[i]     = (bf16_t)(a.x / (1.f + __expf(-a.x)) * b.x);
    act[i + 1] = (bf16_t)(a.y / (1.f + __expf(-a.y)) * b.y);
    act[i + 2] = (bf16_t)(a.z / (1.f + __expf(-a.z)) * b.z);
    act[i + 3] = (bf16_t)(a.w / (1.f + __expf(-a.w)) * b.w);
}

// ---------------------------------------------------------------------------
// Orchestration
// ---------------------------------------------------------------------------
static inline size_t align256(size_t x) { return (x + 255) & ~(size_t)255; }

extern "C" void kernel_launch(void* const* d_in, const int* in_sizes, int n_in,
                              void* d_out, int out_size, void* d_ws, size_t ws_size,
                              hipStream_t stream)
{
    const int*   ids      = (const int*)d_in[0];
    const float* tok_emb  = (const float*)d_in[1];
    const float* pos_emb  = (const float*)d_in[2];
    const float* qkv_w    = (const float*)d_in[3];
    const float* qkv_b    = (const float*)d_in[4];
    const float* out_w    = (const float*)d_in[5];
    const float* out_b    = (const float*)d_in[6];
    const float* ln1_w    = (const float*)d_in[7];
    const float* ln1_b    = (const float*)d_in[8];
    const float* ln2_w    = (const float*)d_in[9];
    const float* ln2_b    = (const float*)d_in[10];
    const float* router_w = (const float*)d_in[11];
    const float* w1       = (const float*)d_in[12];
    const float* w3       = (const float*)d_in[13];
    const float* w2       = (const float*)d_in[14];
    const float* lnf_w    = (const float*)d_in[15];
    const float* lnf_b    = (const float*)d_in[16];
    const float* head_w   = (const float*)d_in[17];
    float* out = (float*)d_out;

    char* p = (char*)d_ws;
    auto take = [&](size_t bytes) { char* r = p; p += align256(bytes); return r; };

    const size_t NQKVW = (size_t)NLAY * 3 * DMODEL * DMODEL;
    const size_t NOUTW = (size_t)NLAY * DMODEL * DMODEL;
    const size_t NEXPW = (size_t)NLAY * NEXP * DFFN * DMODEL;
    const size_t NHEADW = (size_t)VOC * DMODEL;

    bf16_t* wqkvb = (bf16_t*)take(NQKVW * 2);
    bf16_t* woutb = (bf16_t*)take(NOUTW * 2);
    bf16_t* w1b   = (bf16_t*)take(NEXPW * 2);
    bf16_t* w3b   = (bf16_t*)take(NEXPW * 2);
    bf16_t* w2b   = (bf16_t*)take(NEXPW * 2);
    bf16_t* whdb  = (bf16_t*)take(NHEADW * 2);

    float*  x     = (float*)take((size_t)TTOK * DMODEL * 4);
    bf16_t* xb    = (bf16_t*)take((size_t)TTOK * DMODEL * 2);
    float*  qkv   = (float*)take((size_t)TTOK * 3 * DMODEL * 4);
    bf16_t* qkvb  = (bf16_t*)take((size_t)TTOK * 3 * DMODEL * 2);
    bf16_t* vT    = (bf16_t*)take((size_t)DMODEL * TTOK * 2);
    float*  scores= (float*)take((size_t)NHEAD * TTOK * TTOK * 4);
    bf16_t* attb  = (bf16_t*)take((size_t)NHEAD * TTOK * TTOK * 2);
    float*  ctx   = (float*)take((size_t)TTOK * DMODEL * 4);
    bf16_t* ctxb  = (bf16_t*)take((size_t)TTOK * DMODEL * 2);
    float*  attn  = (float*)take((size_t)TTOK * DMODEL * 4);
    float*  y2    = (float*)take((size_t)2 * TTOK * DMODEL * 4);
    int*    topidx= (int*)take((size_t)2 * TTOK * 4);
    float*  topp  = (float*)take((size_t)2 * TTOK * 4);
    int*    tokmap= (int*)take((size_t)GROWS * 4);
    int*    rowmap= (int*)take((size_t)GROWS * 4);
    int*    tileE = (int*)take(32 * 4);

    // Aliases (phase-disjoint): h1/h3 live inside scores; act lives inside qkv
    float*  h1   = scores;                                   // 6.29M f32
    float*  h3   = scores + (size_t)8 * 1024 * 1024;         // 6.29M f32
    bf16_t* actb = (bf16_t*)qkv;                             // 3072*2048 bf16 == S*3D f32 bytes

    auto cvt = [&](const float* src, bf16_t* dst, size_t n) {
        int blocks = (int)((n + 1023) / 1024);
        k_cvt<<<blocks, 256, 0, stream>>>(src, dst, (long long)n);
    };
    auto gemm = [&](const bf16_t* A, const bf16_t* B, float* C, const float* bias,
                    int M, int N, int Kd, int lda, int ldb, int ldc,
                    long long sA, long long sB, long long sC, int Z, float alpha,
                    const int* gA, const int* sCm, const int* tE, long long sBe) {
        dim3 grid((N + 127) / 128, (M + 127) / 128, Z);
        k_gemm_bf16<<<grid, 256, 0, stream>>>(A, B, C, bias, M, N, Kd, lda, ldb, ldc,
                                              sA, sB, sC, alpha, gA, sCm, tE, sBe);
    };

    // --- one-time (per call) weight fp32 -> bf16 ---
    cvt(qkv_w, wqkvb, NQKVW);
    cvt(out_w, woutb, NOUTW);
    cvt(w1, w1b, NEXPW);
    cvt(w3, w3b, NEXPW);
    cvt(w2, w2b, NEXPW);
    cvt(head_w, whdb, NHEADW);

    // --- embedding ---
    k_embed<<<TTOK, 256, 0, stream>>>(ids, tok_emb, pos_emb, x);

    for (int l = 0; l < NLAY; ++l) {
        // QKV projection
        cvt(x, xb, (size_t)TTOK * DMODEL);
        gemm(xb, wqkvb + (size_t)l * 3 * DMODEL * DMODEL, qkv, qkv_b + (size_t)l * 3 * DMODEL,
             TTOK, 3 * DMODEL, DMODEL, DMODEL, DMODEL, 3 * DMODEL,
             0, 0, 0, 1, 1.f, nullptr, nullptr, nullptr, 0);
        cvt(qkv, qkvb, (size_t)TTOK * 3 * DMODEL);
        k_vtrans<<<TTOK, 256, 0, stream>>>(qkv, vT);

        // scores = (Q K^T) / sqrt(64), batched over heads via z-strides
        gemm(qkvb, qkvb + DMODEL, scores, nullptr,
             TTOK, TTOK, DHEAD, 3 * DMODEL, 3 * DMODEL, TTOK,
             DHEAD, DHEAD, (long long)TTOK * TTOK, NHEAD, 0.125f,
             nullptr, nullptr, nullptr, 0);
        k_softmax1024<<<NHEAD * TTOK, 256, 0, stream>>>(scores);
        cvt(scores, attb, (size_t)NHEAD * TTOK * TTOK);

        // ctx = att @ V  (B = V^T per head), writes [t][h*64+n]
        gemm(attb, vT, ctx, nullptr,
             TTOK, DHEAD, TTOK, TTOK, TTOK, DMODEL,
             (long long)TTOK * TTOK, (long long)DHEAD * TTOK, DHEAD, NHEAD, 1.f,
             nullptr, nullptr, nullptr, 0);
        cvt(ctx, ctxb, (size_t)TTOK * DMODEL);

        // output projection + residual LN
        gemm(ctxb, woutb + (size_t)l * DMODEL * DMODEL, attn, out_b + (size_t)l * DMODEL,
             TTOK, DMODEL, DMODEL, DMODEL, DMODEL, DMODEL,
             0, 0, 0, 1, 1.f, nullptr, nullptr, nullptr, 0);
        k_add_ln<<<TTOK, 256, 0, stream>>>(x, attn, ln1_w + (size_t)l * DMODEL,
                                           ln1_b + (size_t)l * DMODEL, x);

        // MoE: route, gather-GEMM experts (top-2 only), scatter, combine-LN
        cvt(x, xb, (size_t)TTOK * DMODEL);
        k_router<<<TTOK, 256, 0, stream>>>(x, router_w + (size_t)l * NEXP * DMODEL,
                                           topidx, topp);
        k_build<<<1, 1, 0, stream>>>(topidx, tokmap, rowmap, tileE);

        gemm(xb, w1b + (size_t)l * NEXP * DFFN * DMODEL, h1, nullptr,
             GROWS, DFFN, DMODEL, DMODEL, DMODEL, DFFN,
             0, 0, 0, 1, 1.f, tokmap, nullptr, tileE, (long long)DFFN * DMODEL);
        gemm(xb, w3b + (size_t)l * NEXP * DFFN * DMODEL, h3, nullptr,
             GROWS, DFFN, DMODEL, DMODEL, DMODEL, DFFN,
             0, 0, 0, 1, 1.f, tokmap, nullptr, tileE, (long long)DFFN * DMODEL);
        k_silu_mul<<<(GROWS * DFFN) / 1024, 256, 0, stream>>>(h1, h3, actb);
        gemm(actb, w2b + (size_t)l * NEXP * DMODEL * DFFN, y2, nullptr,
             GROWS, DMODEL, DFFN, DFFN, DFFN, DMODEL,
             0, 0, 0, 1, 1.f, nullptr, rowmap, tileE, (long long)DMODEL * DFFN);
        k_moe_ln<<<TTOK, 256, 0, stream>>>(x, y2, topp, ln2_w + (size_t)l * DMODEL,
                                           ln2_b + (size_t)l * DMODEL, x);
    }

    // final LN + LM head
    k_add_ln<<<TTOK, 256, 0, stream>>>(x, nullptr, lnf_w, lnf_b, x);
    cvt(x, xb, (size_t)TTOK * DMODEL);
    gemm(xb, whdb, out, nullptr,
         TTOK, VOC, DMODEL, DMODEL, DMODEL, VOC,
         0, 0, 0, 1, 1.f, nullptr, nullptr, nullptr, 0);

    (void)in_sizes; (void)n_in; (void)out_size; (void)ws_size;
}
